// TabRetNet_44633300140439
// MI455X (gfx1250) — compile-verified, matrix operand order checked by reading, not run on hardware
//
#include <hip/hip_runtime.h>
#include <hip/hip_bf16.h>

typedef __attribute__((ext_vector_type(16))) _Float16 v16h;
typedef __attribute__((ext_vector_type(8)))  _Float16 v8h;
typedef __attribute__((ext_vector_type(8)))  float    v8f;
typedef unsigned int v4u __attribute__((ext_vector_type(4)));
typedef int          v4i __attribute__((ext_vector_type(4)));
typedef int          v8i __attribute__((ext_vector_type(8)));

static constexpr int LNUM = 6, DMODEL = 1024, HNUM = 8, FDIM = 2048;
static constexpr int BNUM = 4, SLEN = 2048, INDIM = 128, DKDIM = 128;
static constexpr int MROWS = BNUM * SLEN;       // 8192 token rows

// ---------------------------------------------------------------------------
// Tensor Data Mover: async 2D tile load (global, row-major) -> LDS.
// Descriptor per CDNA5 ISA ch.8: group0 = {count/flags, lds_addr, gaddr, type},
// group1 = {mask/data_size, tensor dims, tile dims, dim0 stride}. 2D tensor ->
// groups 2/3 zero. Issued by one wave; drained with s_wait_tensorcnt.
// ---------------------------------------------------------------------------
#if __has_builtin(__builtin_amdgcn_tensor_load_to_lds) && __has_builtin(__builtin_amdgcn_s_wait_tensorcnt)
#define USE_TDM 1
__device__ __forceinline__ void tdm_load_2d_f16(const void* gptr, unsigned lds_off,
                                                unsigned tile_d0, unsigned tile_d1,
                                                unsigned tensor_d0, unsigned tensor_d1,
                                                unsigned long long stride_d0)
{
    unsigned long long ga = (unsigned long long)(uintptr_t)gptr;
    v4u g0;
    g0[0] = 1u;                                              // count=1 (valid user D#)
    g0[1] = lds_off;                                         // LDS byte address
    g0[2] = (unsigned)(ga & 0xffffffffu);                    // global_addr[31:0]
    g0[3] = (unsigned)((ga >> 32) & 0x1ffffffu) | (2u << 30);// global_addr[56:32] | type=2
    v8i g1;
    g1[0] = (int)(1u << 16);                                 // workgroup_mask=0, data_size=1 (2B)
    g1[1] = (int)((tensor_d0 & 0xffffu) << 16);              // tensor_dim0[15:0]
    g1[2] = (int)(((tensor_d0 >> 16) & 0xffffu) |            // tensor_dim0[31:16]
                  ((tensor_d1 & 0xffffu) << 16));            // tensor_dim1[15:0]
    g1[3] = (int)(((tensor_d1 >> 16) & 0xffffu) |            // tensor_dim1[31:16]
                  (tile_d0 << 16));                          // tile_dim0
    g1[4] = (int)(tile_d1 & 0xffffu);                        // tile_dim1 (tile_dim2=0)
    g1[5] = (int)(stride_d0 & 0xffffffffull);                // tensor_dim0_stride[31:0]
    g1[6] = (int)((stride_d0 >> 32) & 0xffffull);            // stride[47:32], dim1_stride lo=0
    g1[7] = 0;
    v4i zz = {0, 0, 0, 0};
#if __clang_major__ >= 23
    v8i z8 = {0, 0, 0, 0, 0, 0, 0, 0};
    __builtin_amdgcn_tensor_load_to_lds(g0, g1, zz, zz, z8, 0);
#else
    __builtin_amdgcn_tensor_load_to_lds(g0, g1, zz, zz, 0);
#endif
}
#endif

// ---------------------------------------------------------------------------
// fp32 -> fp16 conversion
// ---------------------------------------------------------------------------
__global__ void cvt_f32_f16(const float* __restrict__ a, _Float16* __restrict__ b, size_t n) {
    size_t i = (size_t)blockIdx.x * blockDim.x + threadIdx.x;
    if (i < n) b[i] = (_Float16)a[i];
}

// ---------------------------------------------------------------------------
// WMMA GEMM: C[M,N] = act(A[M,K] @ B[K,N] + bias) (+ residual), f16 in, f32 acc
// Tile 128x128, BK=32, 8 waves: wave grid 4x2, each wave 32x64 (2x4 frags)
// A tile staged via Tensor Data Mover; B tile staged transposed by the waves.
// ---------------------------------------------------------------------------
template<int BIAS, int ACT, int RES, int OUTF16>
__global__ __launch_bounds__(256)
void gemm_wmma(const _Float16* __restrict__ A, const _Float16* __restrict__ Bw,
               const float* __restrict__ bias, const float* __restrict__ res,
               float* __restrict__ Cf, _Float16* __restrict__ Ch,
               int M, int N, int K)
{
    __shared__ _Float16 As[128][32];
    __shared__ _Float16 BsT[128][32];   // [n][k] transposed B tile

    const int tid  = threadIdx.x;
    const int wave = tid >> 5, lane = tid & 31;
    const int tm = blockIdx.y * 128, tn = blockIdx.x * 128;
    const int wm = (wave >> 1) * 32, wn = (wave & 1) * 64;
    const int lm = lane & 15;
    const int kc0 = (lane < 16) ? 0 : 8;
    const int mhalf = (lane < 16) ? 0 : 8;

    v8f acc[2][4];
    const v8f vz = {0.f,0.f,0.f,0.f,0.f,0.f,0.f,0.f};
    for (int i = 0; i < 2; i++) for (int j = 0; j < 4; j++) acc[i][j] = vz;

    for (int k0 = 0; k0 < K; k0 += 32) {
        __syncthreads();               // previous-iteration LDS reads complete
#ifdef USE_TDM
        if (tid < 32) {                // wave 0 issues the async A-tile DMA
            tdm_load_2d_f16(&A[(size_t)tm * K + k0],
                            (unsigned)(uintptr_t)&As[0][0],
                            /*tile*/ 32u, 128u,
                            /*tensor*/ (unsigned)K, (unsigned)M,
                            /*stride*/ (unsigned long long)K);
        }
#else
        for (int i = 0; i < 2; i++) {
            int c = tid + 256 * i;
            int row = c >> 2, kk = (c & 3) * 8;
            *(v8h*)&As[row][kk] = *(const v8h*)&A[(size_t)(tm + row) * K + k0 + kk];
        }
#endif
        // stage B transposed: 32x128 (overlaps the TDM transfer)
        for (int i = 0; i < 2; i++) {
            int c = tid + 256 * i;
            int kr = c >> 4, nc = (c & 15) * 8;
            v8h bv = *(const v8h*)&Bw[(size_t)(k0 + kr) * N + tn + nc];
            for (int j = 0; j < 8; j++) BsT[nc + j][kr] = bv[j];
        }
        if (k0 + 32 < K) {
            __builtin_prefetch(&Bw[(size_t)(k0 + 32 + (tid >> 4)) * N + tn + (tid & 15) * 8], 0, 3);
        }
#ifdef USE_TDM
        if (tid < 32) __builtin_amdgcn_s_wait_tensorcnt(0);
#endif
        __syncthreads();

        v16h af[2], bf[4];
        for (int i = 0; i < 2; i++) {
            int row = wm + i * 16 + lm;
            union { v16h v; v8h h[2]; } u;
            u.h[0] = *(v8h*)&As[row][kc0];
            u.h[1] = *(v8h*)&As[row][kc0 + 16];
            af[i] = u.v;
        }
        for (int j = 0; j < 4; j++) {
            int col = wn + j * 16 + lm;
            union { v16h v; v8h h[2]; } u;
            u.h[0] = *(v8h*)&BsT[col][kc0];
            u.h[1] = *(v8h*)&BsT[col][kc0 + 16];
            bf[j] = u.v;
        }
        for (int i = 0; i < 2; i++)
            for (int j = 0; j < 4; j++)
                acc[i][j] = __builtin_amdgcn_wmma_f32_16x16x32_f16(
                    false, af[i], false, bf[j], (short)0, acc[i][j], false, false);
    }

    for (int i = 0; i < 2; i++) for (int j = 0; j < 4; j++) {
        int col = tn + wn + j * 16 + lm;
        float bvn = BIAS ? bias[col] : 0.f;
        for (int r = 0; r < 8; r++) {
            int row = tm + wm + i * 16 + mhalf + r;
            float v = acc[i][j][r] + bvn;
            if (ACT == 1) {  // tanh-approx gelu
                float t = 0.7978845608028654f * (v + 0.044715f * v * v * v);
                v = 0.5f * v * (1.0f + tanhf(t));
            }
            size_t idx = (size_t)row * N + col;
            if (RES) v += res[idx];
            if (OUTF16) Ch[idx] = (_Float16)v;
            else        Cf[idx] = v;
        }
    }
}

// ---------------------------------------------------------------------------
// LayerNorm over D=1024, fp16 output (feeds WMMA GEMMs)
// ---------------------------------------------------------------------------
__global__ __launch_bounds__(256)
void layernorm_f16(const float* __restrict__ x, const float* __restrict__ s,
                   const float* __restrict__ b, _Float16* __restrict__ y)
{
    const int row = blockIdx.x, tid = threadIdx.x;
    const float* xr = x + (size_t)row * DMODEL;
    float vals[4], lsum = 0.f, lsq = 0.f;
    for (int i = 0; i < 4; i++) {
        float v = xr[tid + 256 * i];
        vals[i] = v; lsum += v; lsq += v * v;
    }
    for (int off = 16; off > 0; off >>= 1) {
        lsum += __shfl_down(lsum, off, 32);
        lsq  += __shfl_down(lsq,  off, 32);
    }
    __shared__ float rs[8], rq[8];
    if ((tid & 31) == 0) { rs[tid >> 5] = lsum; rq[tid >> 5] = lsq; }
    __syncthreads();
    float ts = 0.f, tq = 0.f;
    for (int w = 0; w < 8; w++) { ts += rs[w]; tq += rq[w]; }
    float mu  = ts / (float)DMODEL;
    float inv = rsqrtf(tq / (float)DMODEL - mu * mu + 1e-6f);
    for (int i = 0; i < 4; i++) {
        int d = tid + 256 * i;
        y[(size_t)row * DMODEL + d] = (_Float16)((vals[i] - mu) * inv * s[d] + b[d]);
    }
}

// ---------------------------------------------------------------------------
// RoPE + k-scale + [B,S,H,DK] -> [B,H,S,DK] transpose, v pass-through
// ---------------------------------------------------------------------------
__global__ void rope_kernel(const _Float16* __restrict__ q, const _Float16* __restrict__ k,
                            const _Float16* __restrict__ v, _Float16* __restrict__ qo,
                            _Float16* __restrict__ ko, _Float16* __restrict__ vo)
{
    size_t idx = (size_t)blockIdx.x * blockDim.x + threadIdx.x;  // B*S*H*64
    int d2 = (int)(idx & 63);
    size_t t = idx >> 6;
    int hh = (int)(t & 7);
    size_t bs = t >> 3;
    int s = (int)(bs & (SLEN - 1));
    int b = (int)(bs >> 11);
    float ang = __expf(-9.210340371976184f * (float)d2 / 63.0f);  // 10000^(-d2/63)
    float sn, cs;
    __sincosf((float)s * ang, &sn, &cs);
    size_t in_idx  = bs * DMODEL + (size_t)hh * DKDIM + (size_t)d2 * 2;
    size_t out_idx = (((size_t)(b * HNUM + hh) * SLEN + s) * DKDIM) + (size_t)d2 * 2;
    float q0 = (float)q[in_idx], q1 = (float)q[in_idx + 1];
    float k0 = (float)k[in_idx], k1 = (float)k[in_idx + 1];
    const float ksc = 0.08838834764831845f;  // 128^-0.5
    qo[out_idx]     = (_Float16)(q0 * cs - q1 * sn);
    qo[out_idx + 1] = (_Float16)(q1 * cs + q0 * sn);
    ko[out_idx]     = (_Float16)((k0 * cs - k1 * sn) * ksc);
    ko[out_idx + 1] = (_Float16)((k1 * cs + k0 * sn) * ksc);
    vo[out_idx]     = v[in_idx];
    vo[out_idx + 1] = v[in_idx + 1];
}

// ---------------------------------------------------------------------------
// Retention: per (b,h, 128-row query block). Flash-style over 32-wide key
// blocks: scores via WMMA, decay mask + abs-sum, then score(f16) @ V via WMMA.
// Epilogue: /clip(abssum,1), RMS-norm over DK, * gn_g. Output [B,S,D] fp32.
// ---------------------------------------------------------------------------
__global__ __launch_bounds__(256)
void retention_kernel(const _Float16* __restrict__ q, const _Float16* __restrict__ k,
                      const _Float16* __restrict__ v, const float* __restrict__ gn,
                      float* __restrict__ o)
{
    const int bh = blockIdx.x;      // b*H + h
    const int hh = bh & 7;
    const int qb = blockIdx.y;      // query block of 128 rows
    const int tid = threadIdx.x, wave = tid >> 5, lane = tid & 31;
    const int lm = lane & 15;
    const int kc0 = (lane < 16) ? 0 : 8;
    const int mhalf = (lane < 16) ? 0 : 8;
    const int r0 = qb * 128 + wave * 16;

    __shared__ _Float16 Vt[128][32];         // [d][m] transposed V block
    __shared__ _Float16 Sst[8][16][32];      // per-wave score staging

    const _Float16* qbase = q + (size_t)bh * SLEN * DKDIM;
    const _Float16* kbase = k + (size_t)bh * SLEN * DKDIM;
    const _Float16* vbase = v + (size_t)bh * SLEN * DKDIM;

    const float gamma = 1.f - exp2f(-5.f - (float)hh);
    const float l2g = log2f(gamma);

    float dn[8];
    for (int r = 0; r < 8; r++) {
        float n = (float)(r0 + mhalf + r);
        float ssum = (1.f - exp2f((n + 1.f) * l2g)) / (1.f - gamma);
        dn[r] = rsqrtf(ssum);
    }

    // Q fragments for this wave's 16 rows (DK=128 -> 4 K-steps of 32)
    v16h aq[4];
    {
        const _Float16* qp = qbase + (size_t)(r0 + lm) * DKDIM;
        for (int ks = 0; ks < 4; ks++) {
            union { v16h vv; v8h h[2]; } u;
            u.h[0] = *(const v8h*)(qp + ks * 32 + kc0);
            u.h[1] = *(const v8h*)(qp + ks * 32 + kc0 + 16);
            aq[ks] = u.vv;
        }
    }

    const v8f vzero = {0.f,0.f,0.f,0.f,0.f,0.f,0.f,0.f};
    v8f oacc[8];
    for (int j = 0; j < 8; j++) oacc[j] = vzero;
    float rabs[8] = {0.f,0.f,0.f,0.f,0.f,0.f,0.f,0.f};

    const int nkb = (qb + 1) * 4;            // causal: key blocks of 32
    for (int kb = 0; kb < nkb; kb++) {
        const int m0 = kb * 32;
        __syncthreads();
        // stage V block transposed into LDS
        for (int i = 0; i < 2; i++) {
            int c = tid + 256 * i;
            int mr = c >> 4, nc = (c & 15) * 8;
            v8h vv = *(const v8h*)(vbase + (size_t)(m0 + mr) * DKDIM + nc);
            for (int j = 0; j < 8; j++) Vt[nc + j][mr] = vv[j];
        }
        __syncthreads();

        // scores: 16x32 per wave = 2 accumulators
        v8f sacc[2] = {vzero, vzero};
        for (int ks = 0; ks < 4; ks++) {
            for (int jf = 0; jf < 2; jf++) {
                const _Float16* kp = kbase + (size_t)(m0 + jf * 16 + lm) * DKDIM + ks * 32;
                union { v16h vv; v8h h[2]; } u;
                u.h[0] = *(const v8h*)(kp + kc0);
                u.h[1] = *(const v8h*)(kp + kc0 + 16);
                sacc[jf] = __builtin_amdgcn_wmma_f32_16x16x32_f16(
                    false, aq[ks], false, u.vv, (short)0, sacc[jf], false, false);
            }
        }
        // decay mask + abs-sum + stage to LDS as f16
        for (int jf = 0; jf < 2; jf++) {
            for (int r = 0; r < 8; r++) {
                int n = r0 + mhalf + r;
                int m = m0 + jf * 16 + lm;
                float w = (n >= m) ? exp2f((float)(n - m) * l2g) * dn[r] : 0.f;
                float val = sacc[jf][r] * w;
                rabs[r] += fabsf(val);
                Sst[wave][mhalf + r][jf * 16 + lm] = (_Float16)val;
            }
        }
        __syncthreads();   // order LDS write->read (cross-lane via LDS)

        // read scores back as A fragment, multiply by V block
        union { v16h vv; v8h h[2]; } su;
        su.h[0] = *(v8h*)&Sst[wave][lm][kc0];
        su.h[1] = *(v8h*)&Sst[wave][lm][kc0 + 16];
        for (int jn = 0; jn < 8; jn++) {
            int d = jn * 16 + lm;
            union { v16h vv; v8h h[2]; } bu;
            bu.h[0] = *(v8h*)&Vt[d][kc0];
            bu.h[1] = *(v8h*)&Vt[d][kc0 + 16];
            oacc[jn] = __builtin_amdgcn_wmma_f32_16x16x32_f16(
                false, su.vv, false, bu.vv, (short)0, oacc[jn], false, false);
        }
    }

    // abs-sum: reduce across the 16 lanes holding each row
    for (int r = 0; r < 8; r++)
        for (int off = 1; off < 16; off <<= 1)
            rabs[r] += __shfl_xor(rabs[r], off, 32);

    // 1/clip + collect sum of squares per row
    float sq[8];
    for (int r = 0; r < 8; r++) {
        float inv = 1.f / fmaxf(rabs[r], 1.f);
        float s2 = 0.f;
        for (int jn = 0; jn < 8; jn++) {
            float t = oacc[jn][r] * inv;
            oacc[jn][r] = t;
            s2 += t * t;
        }
        sq[r] = s2;
    }
    for (int r = 0; r < 8; r++)
        for (int off = 1; off < 16; off <<= 1)
            sq[r] += __shfl_xor(sq[r], off, 32);

    const int b = bh >> 3;
    for (int r = 0; r < 8; r++) {
        int srow = r0 + mhalf + r;
        float scale = rsqrtf(sq[r] / (float)DKDIM + 1e-6f);
        float* orow = o + ((size_t)(b * SLEN + srow) * DMODEL) + hh * DKDIM;
        for (int jn = 0; jn < 8; jn++) {
            int d = jn * 16 + lm;
            orow[d] = oacc[jn][r] * scale * gn[hh * DKDIM + d];
        }
    }
}

// ---------------------------------------------------------------------------
// u = silu(g) * o  (fp16 out, feeds Wo GEMM)
// ---------------------------------------------------------------------------
__global__ void gate_kernel(const float* __restrict__ g, const float* __restrict__ o,
                            _Float16* __restrict__ u, size_t n)
{
    size_t i = (size_t)blockIdx.x * blockDim.x + threadIdx.x;
    if (i < n) {
        float gv = g[i];
        float sv = gv / (1.f + __expf(-gv));
        u[i] = (_Float16)(sv * o[i]);
    }
}

// ---------------------------------------------------------------------------
// Final: LN + dot(W_reg) + scaled sigmoid
// ---------------------------------------------------------------------------
__global__ __launch_bounds__(256)
void head_kernel(const float* __restrict__ h, const float* __restrict__ s,
                 const float* __restrict__ b, const float* __restrict__ wr,
                 const float* __restrict__ br, float* __restrict__ out)
{
    const int row = blockIdx.x, tid = threadIdx.x;
    const float* xr = h + (size_t)row * DMODEL;
    float vals[4], lsum = 0.f, lsq = 0.f;
    for (int i = 0; i < 4; i++) {
        float v = xr[tid + 256 * i];
        vals[i] = v; lsum += v; lsq += v * v;
    }
    for (int off = 16; off > 0; off >>= 1) {
        lsum += __shfl_down(lsum, off, 32);
        lsq  += __shfl_down(lsq,  off, 32);
    }
    __shared__ float rs[8], rq[8], rd[8];
    if ((tid & 31) == 0) { rs[tid >> 5] = lsum; rq[tid >> 5] = lsq; }
    __syncthreads();
    float ts = 0.f, tq = 0.f;
    for (int w = 0; w < 8; w++) { ts += rs[w]; tq += rq[w]; }
    float mu  = ts / (float)DMODEL;
    float inv = rsqrtf(tq / (float)DMODEL - mu * mu + 1e-6f);
    float dot = 0.f;
    for (int i = 0; i < 4; i++) {
        int d = tid + 256 * i;
        dot += ((vals[i] - mu) * inv * s[d] + b[d]) * wr[d];
    }
    for (int off = 16; off > 0; off >>= 1) dot += __shfl_down(dot, off, 32);
    if ((tid & 31) == 0) rd[tid >> 5] = dot;
    __syncthreads();
    if (tid == 0) {
        float td = 0.f;
        for (int w = 0; w < 8; w++) td += rd[w];
        out[row] = 1.1f / (1.f + __expf(-(td + br[0]))) - 0.05f;
    }
}

// ---------------------------------------------------------------------------
extern "C" void kernel_launch(void* const* d_in, const int* in_sizes, int n_in,
                              void* d_out, int out_size, void* d_ws, size_t ws_size,
                              hipStream_t stream)
{
    const float* x    = (const float*)d_in[0];
    const float* W_in = (const float*)d_in[1];
    const float* b_in = (const float*)d_in[2];
    const float* Wq   = (const float*)d_in[3];
    const float* Wk   = (const float*)d_in[4];
    const float* Wv   = (const float*)d_in[5];
    const float* Wg   = (const float*)d_in[6];
    const float* Wo   = (const float*)d_in[7];
    const float* gn_g = (const float*)d_in[8];
    const float* ln1_s= (const float*)d_in[9];
    const float* ln1_b= (const float*)d_in[10];
    const float* W1   = (const float*)d_in[11];
    const float* b1   = (const float*)d_in[12];
    const float* W2   = (const float*)d_in[13];
    const float* b2   = (const float*)d_in[14];
    const float* ln2_s= (const float*)d_in[15];
    const float* ln2_b= (const float*)d_in[16];
    const float* lnf_s= (const float*)d_in[17];
    const float* lnf_b= (const float*)d_in[18];
    const float* Wreg = (const float*)d_in[19];
    const float* breg = (const float*)d_in[20];
    float* out = (float*)d_out;

    char* ws = (char*)d_ws;
    size_t off = 0;
    auto alloc = [&](size_t bytes) -> char* {
        char* p = ws + off;
        off += (bytes + 255) & ~(size_t)255;
        return p;
    };

    const size_t MD = (size_t)MROWS * DMODEL;
    float*    h_buf = (float*)   alloc(MD * 4);
    _Float16* y16   = (_Float16*)alloc(MD * 2);
    _Float16* qr16  = (_Float16*)alloc(MD * 2);
    _Float16* kr16  = (_Float16*)alloc(MD * 2);
    _Float16* vr16  = (_Float16*)alloc(MD * 2);
    float*    g_buf = (float*)   alloc(MD * 4);
    _Float16* q16   = (_Float16*)alloc(MD * 2);
    _Float16* k16   = (_Float16*)alloc(MD * 2);
    _Float16* v16   = (_Float16*)alloc(MD * 2);
    float*    o_buf = (float*)   alloc(MD * 4);
    _Float16* u16   = (_Float16*)alloc(MD * 2);
    _Float16* t16   = (_Float16*)alloc((size_t)MROWS * FDIM * 2);
    _Float16* x16   = (_Float16*)alloc((size_t)MROWS * INDIM * 2);
    _Float16* win16 = (_Float16*)alloc((size_t)INDIM * DMODEL * 2);
    const size_t WDD = (size_t)LNUM * DMODEL * DMODEL;
    _Float16* wq16  = (_Float16*)alloc(WDD * 2);
    _Float16* wk16  = (_Float16*)alloc(WDD * 2);
    _Float16* wv16  = (_Float16*)alloc(WDD * 2);
    _Float16* wg16  = (_Float16*)alloc(WDD * 2);
    _Float16* wo16  = (_Float16*)alloc(WDD * 2);
    const size_t WDF = (size_t)LNUM * DMODEL * FDIM;
    _Float16* w116  = (_Float16*)alloc(WDF * 2);
    _Float16* w216  = (_Float16*)alloc(WDF * 2);

    auto cvt = [&](const float* src, _Float16* dst, size_t n) {
        cvt_f32_f16<<<dim3((unsigned)((n + 255) / 256)), dim3(256), 0, stream>>>(src, dst, n);
    };
    cvt(x,    x16,   (size_t)MROWS * INDIM);
    cvt(W_in, win16, (size_t)INDIM * DMODEL);
    cvt(Wq, wq16, WDD);  cvt(Wk, wk16, WDD);  cvt(Wv, wv16, WDD);
    cvt(Wg, wg16, WDD);  cvt(Wo, wo16, WDD);
    cvt(W1, w116, WDF);  cvt(W2, w216, WDF);

    const dim3 blk(256);
    const dim3 gD (DMODEL / 128, MROWS / 128);   // N=1024
    const dim3 gF (FDIM   / 128, MROWS / 128);   // N=2048

    // h = x @ W_in + b_in
    gemm_wmma<1,0,0,0><<<gD, blk, 0, stream>>>(x16, win16, b_in, nullptr,
                                               h_buf, nullptr, MROWS, DMODEL, INDIM);

    for (int l = 0; l < LNUM; l++) {
        const size_t wdd = (size_t)l * DMODEL * DMODEL;
        const size_t wdf = (size_t)l * DMODEL * FDIM;

        layernorm_f16<<<MROWS, blk, 0, stream>>>(h_buf, ln1_s + l * DMODEL,
                                                 ln1_b + l * DMODEL, y16);
        gemm_wmma<0,0,0,1><<<gD, blk, 0, stream>>>(y16, wq16 + wdd, nullptr, nullptr,
                                                   nullptr, qr16, MROWS, DMODEL, DMODEL);
        gemm_wmma<0,0,0,1><<<gD, blk, 0, stream>>>(y16, wk16 + wdd, nullptr, nullptr,
                                                   nullptr, kr16, MROWS, DMODEL, DMODEL);
        gemm_wmma<0,0,0,1><<<gD, blk, 0, stream>>>(y16, wv16 + wdd, nullptr, nullptr,
                                                   nullptr, vr16, MROWS, DMODEL, DMODEL);
        gemm_wmma<0,0,0,0><<<gD, blk, 0, stream>>>(y16, wg16 + wdd, nullptr, nullptr,
                                                   g_buf, nullptr, MROWS, DMODEL, DMODEL);
        rope_kernel<<<dim3((unsigned)(MD / 2 / 256)), blk, 0, stream>>>(
            qr16, kr16, vr16, q16, k16, v16);
        retention_kernel<<<dim3(BNUM * HNUM, SLEN / 128), blk, 0, stream>>>(
            q16, k16, v16, gn_g + l * DMODEL, o_buf);
        gate_kernel<<<dim3((unsigned)(MD / 256)), blk, 0, stream>>>(g_buf, o_buf, u16, MD);
        gemm_wmma<0,0,1,0><<<gD, blk, 0, stream>>>(u16, wo16 + wdd, nullptr, h_buf,
                                                   h_buf, nullptr, MROWS, DMODEL, DMODEL);

        layernorm_f16<<<MROWS, blk, 0, stream>>>(h_buf, ln2_s + l * DMODEL,
                                                 ln2_b + l * DMODEL, y16);
        gemm_wmma<1,1,0,1><<<gF, blk, 0, stream>>>(y16, w116 + wdf, b1 + l * FDIM, nullptr,
                                                   nullptr, t16, MROWS, FDIM, DMODEL);
        gemm_wmma<1,0,1,0><<<gD, blk, 0, stream>>>(t16, w216 + wdf, b2 + l * DMODEL, h_buf,
                                                   h_buf, nullptr, MROWS, DMODEL, FDIM);
    }

    head_kernel<<<MROWS, blk, 0, stream>>>(h_buf, lnf_s, lnf_b, Wreg, breg, out);
}
